// AttributeAwareCrossAttention_68049461838590
// MI455X (gfx1250) — matrix operand order repaired and checked by reading, not compiled, not run on hardware
//
#include <hip/hip_runtime.h>
#include <hip/hip_bf16.h>

// MI455X (gfx1250) attribute-aware cross attention, flash style.
//   proj_kernel : Q = Wq*x+bq -> bf16 [B][N][C]
//                 K = Wk*a+bk -> bf16 [B][N][C]
//                 V = Wv*a+bv -> bf16 [B][C][N]
//   flash_kernel: per (b, 64-query tile): async-stage 64-key tile in LDS,
//                 S = Q^T K (wmma bf16), online softmax, O += V P^T (wmma bf16),
//                 epilogue O/l + x residual.

#define CDIM 256
#define NPIX 4096
#define BATCH 8

typedef __attribute__((ext_vector_type(16))) __bf16       v16bf;
typedef __attribute__((ext_vector_type(2)))  __bf16       v2bf;
typedef __attribute__((ext_vector_type(8)))  float        v8f;
typedef __attribute__((ext_vector_type(8)))  unsigned int v8u;
typedef __attribute__((ext_vector_type(4)))  unsigned int u32x4;
typedef __attribute__((ext_vector_type(4)))  int          i32x4;

#define AS1 __attribute__((address_space(1)))
#define AS3 __attribute__((address_space(3)))

#if __has_builtin(__builtin_amdgcn_global_load_async_to_lds_b128) && \
    __has_builtin(__builtin_amdgcn_s_wait_asynccnt)
#define USE_ASYNC 1
#else
#define USE_ASYNC 0
#endif

__device__ __forceinline__ unsigned short f2bf(float f) {
    unsigned int u = __builtin_bit_cast(unsigned int, f);
    u += 0x7fffu + ((u >> 16) & 1u);          // round-to-nearest-even
    return (unsigned short)(u >> 16);
}
#if __has_builtin(__builtin_amdgcn_cvt_pk_bf16_f32)
__device__ __forceinline__ unsigned int pk2(float a, float b) {
    v2bf r = __builtin_amdgcn_cvt_pk_bf16_f32(a, b);
    return __builtin_bit_cast(unsigned int, r);
}
#else
// builtin not declared on this toolchain; the ISA has the op -> inline asm.
__device__ __forceinline__ unsigned int pk2(float a, float b) {
    unsigned int r;
    asm("v_cvt_pk_bf16_f32 %0, %1, %2" : "=v"(r) : "v"(a), "v"(b));
    return r;
}
#endif

__device__ __forceinline__ void async_copy16(const unsigned short* g, unsigned short* l) {
#if USE_ASYNC
    __builtin_amdgcn_global_load_async_to_lds_b128(
        (AS1 i32x4*)g, (AS3 i32x4*)l, 0, 0);
#else
    *(u32x4*)l = *(const u32x4*)g;
#endif
}
__device__ __forceinline__ void async_wait() {
#if USE_ASYNC
    __builtin_amdgcn_s_wait_asynccnt(0);
#endif
}

__device__ __forceinline__ v8f wmma_bf16(v16bf a, v16bf b, v8f c) {
    return __builtin_amdgcn_wmma_f32_16x16x32_bf16(
        /*neg_a=*/false, a, /*neg_b=*/false, b,
        /*c_mod=*/(short)0, c, /*reuse_a=*/false, /*reuse_b=*/false);
}

// ---- fragment loaders -------------------------------------------------------
// A 16x32 bf16: lane row m = lane&15; lanes 0-15 hold K 0-7 & 16-23,
//               lanes 16-31 hold K 8-15 & 24-31 (pairs per VGPR).
__device__ __forceinline__ v16bf ldg_a(const unsigned short* rowp, int lane) {
    const int khalf = (lane >> 4) << 3;
    u32x4 lo = *(const u32x4*)(rowp + khalf);
    u32x4 hi = *(const u32x4*)(rowp + 16 + khalf);
    v8u a; a[0]=lo[0]; a[1]=lo[1]; a[2]=lo[2]; a[3]=lo[3];
           a[4]=hi[0]; a[5]=hi[1]; a[6]=hi[2]; a[7]=hi[3];
    return __builtin_bit_cast(v16bf, a);
}
__device__ __forceinline__ v16bf lds_a(const unsigned short* rowp, int lane) {
    const unsigned int* p = (const unsigned int*)rowp;
    const int kh = (lane >> 4) << 2;          // khalf/2 uints
    v8u a; a[0]=p[kh+0]; a[1]=p[kh+1]; a[2]=p[kh+2]; a[3]=p[kh+3];
           a[4]=p[8+kh+0]; a[5]=p[8+kh+1]; a[6]=p[8+kh+2]; a[7]=p[8+kh+3];
    return __builtin_bit_cast(v16bf, a);
}
// B 32x16 bf16: lane col n = lane&15; lanes 0-15 hold K 0-7 & 16-23, 16-31 hold K 8-15 & 24-31.
__device__ __forceinline__ v16bf lds_b(const unsigned short* rowp, int lane) {
    const unsigned int* p = (const unsigned int*)rowp;
    const int kb = (lane >> 4) << 2;          // khalf/2 uints
    v8u a;
#pragma unroll
    for (int i = 0; i < 8; ++i) a[i] = p[(i < 4 ? 0 : 8) + kb + (i & 3)];
    return __builtin_bit_cast(v16bf, a);
}

// ---- QKV projection ---------------------------------------------------------
// grid (NPIX/64, CDIM/64, BATCH*3); block 256 (8 waves).
__global__ __launch_bounds__(256) void proj_kernel(
    const float* __restrict__ x, const float* __restrict__ attr,
    const float* __restrict__ Wq, const float* __restrict__ bq,
    const float* __restrict__ Wk, const float* __restrict__ bk,
    const float* __restrict__ Wv, const float* __restrict__ bv,
    unsigned short* __restrict__ Qbf, unsigned short* __restrict__ Kbf,
    unsigned short* __restrict__ Vbf)
{
    __shared__ unsigned short Wlds[64][CDIM]; // 32 KB weight tile (bf16, staged once)
    __shared__ unsigned short Xlds[64][32];   // [pixel][cin-chunk], transposed stage

    const int tid  = threadIdx.x;
    const int lane = tid & 31;
    const int wid  = tid >> 5;
    const int n0   = blockIdx.x * 64;         // pixel tile
    const int m0   = blockIdx.y * 64;         // cout tile
    const int which = blockIdx.z % 3;         // 0=q 1=k 2=v
    const int b     = blockIdx.z / 3;

    const float* X    = ((which == 0) ? x : attr) + (size_t)b * CDIM * NPIX;
    const float* Wm   = (which == 0) ? Wq : (which == 1) ? Wk : Wv;
    const float* bias = (which == 0) ? bq : (which == 1) ? bk : bv;

    const int m_base = (wid & 3) * 16;
    const int n_base = (wid >> 2) * 32;
    const int half   = lane >> 4;
    const int l15    = lane & 15;

    {   // stage 64x256 f32 weight tile -> bf16 LDS, converted exactly once
        const int row = tid >> 2;             // 64 rows, 4 threads/row
        const int cb  = (tid & 3) * 64;       // 64 f32 each
        const float* src = Wm + (size_t)(m0 + row) * CDIM + cb;
        unsigned int* dst = (unsigned int*)&Wlds[row][cb];
#pragma unroll
        for (int i = 0; i < 16; ++i) {
            float4 f = *(const float4*)(src + i * 4);
            dst[i * 2 + 0] = pk2(f.x, f.y);
            dst[i * 2 + 1] = pk2(f.z, f.w);
        }
    }

    v8f acc0 = {}; v8f acc1 = {};

    for (int c0 = 0; c0 < CDIM; c0 += 32) {
        __syncthreads();
        {   // stage X tile transposed: Xlds[n][k] = X[c0+k][n0+n]
            const int k = tid >> 3;
            const int n = (tid & 7) * 8;
            const float* src = X + (size_t)(c0 + k) * NPIX + n0 + n;
            float4 f0 = *(const float4*)(src);
            float4 f1 = *(const float4*)(src + 4);
            Xlds[n+0][k] = f2bf(f0.x); Xlds[n+1][k] = f2bf(f0.y);
            Xlds[n+2][k] = f2bf(f0.z); Xlds[n+3][k] = f2bf(f0.w);
            Xlds[n+4][k] = f2bf(f1.x); Xlds[n+5][k] = f2bf(f1.y);
            Xlds[n+6][k] = f2bf(f1.z); Xlds[n+7][k] = f2bf(f1.w);
        }
        __syncthreads();
        if (which < 2) {
            // D[m=cout][n=pixel]; A = W tile (LDS), B = X^T (LDS)
            v16bf a  = lds_a(&Wlds[m_base + l15][c0], lane);
            v16bf b0 = lds_b(&Xlds[n_base +      l15][0], lane);
            v16bf b1 = lds_b(&Xlds[n_base + 16 + l15][0], lane);
            acc0 = wmma_bf16(a, b0, acc0);
            acc1 = wmma_bf16(a, b1, acc1);
        } else {
            // D[m=pixel][n=cout]; A = X^T (LDS), B = W tile (LDS)
            v16bf a  = lds_a(&Xlds[m_base + l15][0], lane);
            v16bf b0 = lds_b(&Wlds[n_base +      l15][c0], lane);
            v16bf b1 = lds_b(&Wlds[n_base + 16 + l15][c0], lane);
            acc0 = wmma_bf16(a, b0, acc0);
            acc1 = wmma_bf16(a, b1, acc1);
        }
    }

    if (which < 2) {
        // store to [B][N][C]: lane = pixel, rows r = contiguous cout -> 16B store
        unsigned short* Out = (which == 0) ? Qbf : Kbf;
        float4 bv0 = *(const float4*)(bias + m0 + m_base + 8 * half);
        float4 bv1 = *(const float4*)(bias + m0 + m_base + 8 * half + 4);
        float bb[8] = {bv0.x, bv0.y, bv0.z, bv0.w, bv1.x, bv1.y, bv1.z, bv1.w};
#pragma unroll
        for (int sub = 0; sub < 2; ++sub) {
            v8f acc = sub ? acc1 : acc0;
            const int pixel = n0 + n_base + sub * 16 + l15;
            unsigned short* dst =
                Out + ((size_t)b * NPIX + pixel) * CDIM + m0 + m_base + 8 * half;
            u32x4 w;
            w[0] = pk2(acc[0] + bb[0], acc[1] + bb[1]);
            w[1] = pk2(acc[2] + bb[2], acc[3] + bb[3]);
            w[2] = pk2(acc[4] + bb[4], acc[5] + bb[5]);
            w[3] = pk2(acc[6] + bb[6], acc[7] + bb[7]);
            *(u32x4*)dst = w;
        }
    } else {
        // store to [B][C][N]: lane = cout, rows r = contiguous pixel -> 16B store
#pragma unroll
        for (int sub = 0; sub < 2; ++sub) {
            v8f acc = sub ? acc1 : acc0;
            const int cout = m0 + n_base + sub * 16 + l15;
            const float bbv = bias[cout];
            unsigned short* dst =
                Vbf + ((size_t)b * CDIM + cout) * NPIX + n0 + m_base + 8 * half;
            u32x4 w;
            w[0] = pk2(acc[0] + bbv, acc[1] + bbv);
            w[1] = pk2(acc[2] + bbv, acc[3] + bbv);
            w[2] = pk2(acc[4] + bbv, acc[5] + bbv);
            w[3] = pk2(acc[6] + bbv, acc[7] + bbv);
            *(u32x4*)dst = w;
        }
    }
}

// ---- flash attention --------------------------------------------------------
// grid (NPIX/64, BATCH); block 256 (8 waves).
// Wave w: S subtiles (i_sub = w&3, j half = w>>2); O rows c in [w*32, w*32+32).
__global__ __launch_bounds__(256) void flash_kernel(
    const unsigned short* __restrict__ Qbf, const unsigned short* __restrict__ Kbf,
    const unsigned short* __restrict__ Vbf, const float* __restrict__ x,
    float* __restrict__ out)
{
    __shared__ unsigned short Klds[64][CDIM];   // 32 KB key tile (async staged)
    __shared__ float          Slds[64][64];     // 16 KB scores
    __shared__ unsigned short Plds[64][64];     //  8 KB probabilities (bf16)
    __shared__ float          mstate[64], lstate[64], fac[64];

    const int tid  = threadIdx.x;
    const int lane = tid & 31;
    const int wid  = tid >> 5;
    const int i0   = blockIdx.x * 64;
    const int b    = blockIdx.y;
    const int l15  = lane & 15;
    const int half = lane >> 4;

    const unsigned short* Qb = Qbf + (size_t)b * NPIX * CDIM;
    const unsigned short* Kb = Kbf + (size_t)b * NPIX * CDIM;
    const unsigned short* Vb = Vbf + (size_t)b * CDIM * NPIX;

    const int isub_s = wid & 3;       // S i-subtile
    const int jh     = wid >> 2;      // S j-half (32 cols)

    v8f oacc[4][2] = {};              // [i_sub][c_sub]

    if (tid < 64) { mstate[tid] = -1e30f; lstate[tid] = 0.0f; }
    __syncthreads();

    const unsigned short* qrow = Qb + (size_t)(i0 + isub_s * 16 + l15) * CDIM;

    for (int j0 = 0; j0 < NPIX; j0 += 64) {
        // ---- async stage 64x256 bf16 key tile into LDS ----
        {
            const int row  = tid >> 2;            // 64 rows, 4 threads/row
            const int coff = (tid & 3) * 64;      // 64 bf16 elements per thread
            const unsigned short* gsrc = Kb + (size_t)(j0 + row) * CDIM + coff;
            unsigned short* ldst = &Klds[row][coff];
#pragma unroll
            for (int i = 0; i < 8; ++i)
                async_copy16(gsrc + i * 8, ldst + i * 8);
            if (j0 + 64 < NPIX)                   // warm GL2 for next key tile
                __builtin_prefetch(gsrc + (size_t)64 * CDIM, 0, 1);
        }
        async_wait();
        __syncthreads();

        // ---- S = Q^T K for this 64x64 tile ----
        v8f s0 = {}, s1 = {};
#pragma unroll
        for (int c0 = 0; c0 < CDIM; c0 += 32) {
            v16bf aq  = ldg_a(qrow + c0, lane);
            v16bf bk0 = lds_b(&Klds[jh * 32 +      l15][c0], lane);
            v16bf bk1 = lds_b(&Klds[jh * 32 + 16 + l15][c0], lane);
            s0 = wmma_bf16(aq, bk0, s0);
            s1 = wmma_bf16(aq, bk1, s1);
        }
        {   // scatter S to LDS (D layout: lane = j col, vgpr r = i row)
            const int jc = jh * 32 + l15;
            const int ib = isub_s * 16 + 8 * half;
#pragma unroll
            for (int r = 0; r < 8; ++r) {
                Slds[ib + r][jc]      = s0[r];
                Slds[ib + r][jc + 16] = s1[r];
            }
        }
        __syncthreads();

        // ---- online softmax: one thread per query row ----
        if (tid < 64) {
            const float mold = mstate[tid];
            float mx = mold;
#pragma unroll 8
            for (int j = 0; j < 64; ++j) mx = fmaxf(mx, Slds[tid][j]);
            const float f = __expf(mold - mx);
            float sum = 0.0f;
#pragma unroll 4
            for (int j = 0; j < 64; j += 2) {
                const float p0 = __expf(Slds[tid][j]     - mx);
                const float p1 = __expf(Slds[tid][j + 1] - mx);
                sum += p0 + p1;
                *(unsigned int*)&Plds[tid][j] = pk2(p0, p1);
            }
            mstate[tid] = mx;
            lstate[tid] = lstate[tid] * f + sum;
            fac[tid]    = f;
        }
        __syncthreads();

        // ---- rescale accumulators, then O += V * P^T ----
#pragma unroll
        for (int isub = 0; isub < 4; ++isub) {
            const float f = fac[isub * 16 + l15];   // O D-layout: lane = i
#pragma unroll
            for (int cs = 0; cs < 2; ++cs)
#pragma unroll
                for (int r = 0; r < 8; ++r) oacc[isub][cs][r] *= f;
        }
#pragma unroll
        for (int jc = 0; jc < 64; jc += 32) {
            v16bf av0 = ldg_a(Vb + (size_t)(wid * 32 +      l15) * NPIX + j0 + jc, lane);
            v16bf av1 = ldg_a(Vb + (size_t)(wid * 32 + 16 + l15) * NPIX + j0 + jc, lane);
#pragma unroll
            for (int isub = 0; isub < 4; ++isub) {
                v16bf bp = lds_b(&Plds[isub * 16 + l15][jc], lane);
                oacc[isub][0] = wmma_bf16(av0, bp, oacc[isub][0]);
                oacc[isub][1] = wmma_bf16(av1, bp, oacc[isub][1]);
            }
        }
        __syncthreads();
    }

    // ---- epilogue: O/l + residual x ----
    const float* xb = x   + (size_t)b * CDIM * NPIX;
    float*       ob = out + (size_t)b * CDIM * NPIX;
#pragma unroll
    for (int isub = 0; isub < 4; ++isub) {
        const float linv = 1.0f / lstate[isub * 16 + l15];
        const int ig = i0 + isub * 16 + l15;
#pragma unroll
        for (int cs = 0; cs < 2; ++cs) {
            const int cbase = wid * 32 + cs * 16 + 8 * half;
#pragma unroll
            for (int r = 0; r < 8; ++r) {
                const size_t idx = (size_t)(cbase + r) * NPIX + ig;
                ob[idx] = oacc[isub][cs][r] * linv + xb[idx];
            }
        }
    }
}

// ---- launch -----------------------------------------------------------------
extern "C" void kernel_launch(void* const* d_in, const int* in_sizes, int n_in,
                              void* d_out, int out_size, void* d_ws, size_t ws_size,
                              hipStream_t stream) {
    const float* x    = (const float*)d_in[0];
    const float* attr = (const float*)d_in[1];
    const float* Wq   = (const float*)d_in[2];
    const float* bq   = (const float*)d_in[3];
    const float* Wk   = (const float*)d_in[4];
    const float* bk   = (const float*)d_in[5];
    const float* Wv   = (const float*)d_in[6];
    const float* bv   = (const float*)d_in[7];
    float* out = (float*)d_out;

    const size_t planesz = (size_t)BATCH * NPIX * CDIM;   // elements
    unsigned short* Qbf = (unsigned short*)d_ws;
    unsigned short* Kbf = Qbf + planesz;
    unsigned short* Vbf = Kbf + planesz;

    dim3 gp(NPIX / 64, CDIM / 64, BATCH * 3);
    proj_kernel<<<gp, 256, 0, stream>>>(x, attr, Wq, bq, Wk, bk, Wv, bv, Qbf, Kbf, Vbf);

    dim3 gf(NPIX / 64, BATCH);
    flash_kernel<<<gf, 256, 0, stream>>>(Qbf, Kbf, Vbf, x, out);
}
